// CharModel_88811333747032
// MI455X (gfx1250) — matrix operand order repaired and checked by reading, not compiled, not run on hardware
//
#include <hip/hip_runtime.h>
#include <math.h>

typedef __bf16 bf16;
typedef __attribute__((ext_vector_type(16))) __bf16 v16bf;
typedef __attribute__((ext_vector_type(8)))  __bf16 bf16x8;
typedef __attribute__((ext_vector_type(8)))  float  v8f;

#define WMMA_BF16(A, Bm, C) \
  __builtin_amdgcn_wmma_f32_16x16x32_bf16(false, (A), false, (Bm), (short)0, (C), false, false)

__device__ __forceinline__ float fast_tanh(float x) {
#if __has_builtin(__builtin_amdgcn_tanhf)
    return __builtin_amdgcn_tanhf(x);     // native v_tanh_f32 (TRANS op, co-executes with WMMA)
#else
    return tanhf(x);
#endif
}

constexpr int Bsz = 4096, T = 128, V = 128, E = 42, H = 256;
constexpr int BM   = 32;    // batch rows per block
constexpr int NTHR = 512;   // 16 waves

constexpr int AP = H + 8;   // padded pitch (bf16 elems) -> 264 (spreads LDS banks)
constexpr int SP = T + 1;   // seq pitch (ints)  -> 129
constexpr int LP = V + 2;   // logit pitch (f32) -> 130

// LDS layout (bytes)
constexpr int OFF_WTH   = 0;                              // bf16 [H][AP]  (reused as WtOut [V][AP])
constexpr int OFF_XTAB  = OFF_WTH   + H * AP * 2;         // bf16 [V][AP]
constexpr int OFF_ABUF  = OFF_XTAB  + V * AP * 2;         // bf16 [2][BM][AP]  double-buffered
constexpr int OFF_SEQ   = OFF_ABUF  + 2 * BM * AP * 2;    // int  [BM][SP]
constexpr int OFF_LOGIT = OFF_SEQ   + BM * SP * 4;        // f32  [BM][LP]
constexpr int OFF_BH    = OFF_LOGIT + BM * LP * 4;        // f32  [H]
constexpr int OFF_BO    = OFF_BH    + H * 4;              // f32  [V]
constexpr int OFF_RSTAT = OFF_BO    + V * 4;              // f32  [BM][2]
constexpr int SMEM_BYTES = OFF_RSTAT + BM * 2 * 4;        // = 271488 (< 320KB)

union AB16 { bf16x8 h[2]; v16bf v; };

__global__ __launch_bounds__(NTHR)
void rnn_char_model(const int* __restrict__ seq, const float* __restrict__ emb,
                    const float* __restrict__ W_in, const float* __restrict__ b_in,
                    const float* __restrict__ W_h,  const float* __restrict__ b_h,
                    const float* __restrict__ W_out, const float* __restrict__ b_out,
                    float* __restrict__ out)
{
    extern __shared__ char smem[];
    bf16*  wtH   = (bf16*)(smem + OFF_WTH);
    bf16*  xtab  = (bf16*)(smem + OFF_XTAB);
    bf16*  aBuf  = (bf16*)(smem + OFF_ABUF);     // two buffers of BM*AP
    int*   seqB  = (int*) (smem + OFF_SEQ);
    float* logit = (float*)(smem + OFF_LOGIT);
    float* bhS   = (float*)(smem + OFF_BH);
    float* boS   = (float*)(smem + OFF_BO);
    float* rstat = (float*)(smem + OFF_RSTAT);

    const int tid   = threadIdx.x;
    const int lane  = tid & 31;
    const int wave  = tid >> 5;
    const int nlane = lane & 15;
    const int hi    = lane >> 4;       // 0: lanes 0-15, 1: lanes 16-31
    const int bBase = blockIdx.x * BM;

    // ---- Stage W_h (transposed, bf16), biases, sequence rows ----
    for (int i = tid; i < H * H; i += NTHR) {
        int k = i >> 8, n = i & (H - 1);
        wtH[n * AP + k] = (bf16)W_h[i];
    }
    for (int i = tid; i < H; i += NTHR) bhS[i] = b_h[i];
    for (int i = tid; i < V; i += NTHR) boS[i] = b_out[i];
    for (int i = tid; i < BM * T; i += NTHR) {
        int r = i >> 7, t = i & (T - 1);
        seqB[r * SP + t] = seq[(bBase + r) * T + t];
    }
    // ---- XTable[v][h] = relu(emb[v] @ W_in + b_in), bf16 ----
    for (int i = tid; i < V * H; i += NTHR) {
        int v = i >> 8, hc = i & (H - 1);
        float acc = b_in[hc];
        for (int e = 0; e < E; ++e) acc += emb[v * E + e] * W_in[e * H + hc];
        xtab[v * AP + hc] = (bf16)fmaxf(acc, 0.0f);
    }
    __syncthreads();

    // ---- a_0 = x_0 (h_{-1} = 0) into buffer 0 ----
    for (int i = tid; i < BM * H; i += NTHR) {
        int r = i >> 8, c = i & (H - 1);
        aBuf[r * AP + c] = xtab[seqB[r * SP + 0] * AP + c];
    }

    // ---- Persistent B fragments of W_h for this wave's column tile n = wave ----
    const int n0 = wave * 16;
    v16bf bfrag[8];
    {
        const bf16* brow = wtH + (n0 + nlane) * AP;
        #pragma unroll
        for (int kk = 0; kk < 8; ++kk) {
            int kb = kk * 32 + hi * 8;
            AB16 u;
            u.h[0] = *(const bf16x8*)(brow + kb);
            u.h[1] = *(const bf16x8*)(brow + kb + 16);
            bfrag[kk] = u.v;
        }
    }
    const float bhv  = bhS[n0 + nlane];
    const int   colH = n0 + nlane;
    __syncthreads();

    // ---- Recurrence: h_t = tanh(a_t @ W_h + b_h); a_{t+1} = h_t + x_{t+1} ----
    // Double-buffered aBuf: one barrier per step. Depth-1 software pipeline on A frags.
    for (int t = 0; t < T; ++t) {
        const bf16* aCur = aBuf + (t & 1) * (BM * AP);
        bf16*       aNxt = aBuf + ((t + 1) & 1) * (BM * AP);
        const bf16* ar0  = aCur + nlane * AP + hi * 8;
        const bf16* ar1  = aCur + (16 + nlane) * AP + hi * 8;

        v8f acc0 = {};
        v8f acc1 = {};
        AB16 a0c, a1c, a0n, a1n;
        a0c.h[0] = *(const bf16x8*)(ar0);
        a0c.h[1] = *(const bf16x8*)(ar0 + 16);
        a1c.h[0] = *(const bf16x8*)(ar1);
        a1c.h[1] = *(const bf16x8*)(ar1 + 16);
        #pragma unroll
        for (int kk = 0; kk < 8; ++kk) {
            if (kk < 7) {                        // prefetch next K-slab while WMMAs run
                int kA = (kk + 1) * 32;
                a0n.h[0] = *(const bf16x8*)(ar0 + kA);
                a0n.h[1] = *(const bf16x8*)(ar0 + kA + 16);
                a1n.h[0] = *(const bf16x8*)(ar1 + kA);
                a1n.h[1] = *(const bf16x8*)(ar1 + kA + 16);
            }
            acc0 = WMMA_BF16(a0c.v, bfrag[kk], acc0);
            acc1 = WMMA_BF16(a1c.v, bfrag[kk], acc1);
            a0c = a0n;
            a1c = a1n;
        }

        // Epilogue: D layout VGPR i -> (M=i / M=i+8, N=lane)
        if (t + 1 < T) {
            #pragma unroll
            for (int i = 0; i < 8; ++i) {
                int r0 = i + hi * 8;
                float h0 = fast_tanh(acc0[i] + bhv);
                float h1 = fast_tanh(acc1[i] + bhv);
                int tok0 = seqB[r0 * SP + t + 1];
                int tok1 = seqB[(16 + r0) * SP + t + 1];
                aNxt[r0 * AP + colH]        = (bf16)(h0 + (float)xtab[tok0 * AP + colH]);
                aNxt[(16 + r0) * AP + colH] = (bf16)(h1 + (float)xtab[tok1 * AP + colH]);
            }
        } else {   // final step: stash h_T (bf16) for the output GEMM
            #pragma unroll
            for (int i = 0; i < 8; ++i) {
                int r0 = i + hi * 8;
                aNxt[r0 * AP + colH]        = (bf16)fast_tanh(acc0[i] + bhv);
                aNxt[(16 + r0) * AP + colH] = (bf16)fast_tanh(acc1[i] + bhv);
            }
        }
        __syncthreads();   // single barrier per step (double-buffered)
    }

    const bf16* hBuf = aBuf + (T & 1) * (BM * AP);   // where h_T landed

    // ---- Stage W_out transposed bf16 into the (now free) W_h region ----
    for (int i = tid; i < H * V; i += NTHR) {
        int k = i >> 7, v = i & (V - 1);
        wtH[v * AP + k] = (bf16)W_out[i];
    }
    __syncthreads();

    // ---- Output GEMM: logits[32][128] = h @ W_out + b_out (16 tiles, 1/wave) ----
    {
        const int m = wave >> 3, n = wave & 7;
        const bf16* brow = wtH + (n * 16 + nlane) * AP;
        const bf16* arow = hBuf + (m * 16 + nlane) * AP;
        v8f acc = {};
        #pragma unroll
        for (int kk = 0; kk < 8; ++kk) {
            int kA = kk * 32 + hi * 8;
            int kB = kk * 32 + hi * 8;
            AB16 ua, ub;
            ua.h[0] = *(const bf16x8*)(arow + kA);
            ua.h[1] = *(const bf16x8*)(arow + kA + 16);
            ub.h[0] = *(const bf16x8*)(brow + kB);
            ub.h[1] = *(const bf16x8*)(brow + kB + 16);
            acc = WMMA_BF16(ua.v, ub.v, acc);
        }
        float bo = boS[n * 16 + nlane];
        #pragma unroll
        for (int i = 0; i < 8; ++i) {
            int r = m * 16 + i + hi * 8;
            logit[r * LP + n * 16 + nlane] = acc[i] + bo;
        }
    }
    __syncthreads();

    // ---- log-softmax stats (one row per lane of wave 0) ----
    if (tid < BM) {
        const float* row = logit + tid * LP;
        float mx = -INFINITY;
        for (int v = 0; v < V; ++v) mx = fmaxf(mx, row[v]);
        float s = 0.0f;
        for (int v = 0; v < V; ++v) s += __expf(row[v] - mx);
        rstat[tid * 2 + 0] = mx;
        rstat[tid * 2 + 1] = __logf(s);
    }
    __syncthreads();

    // ---- Coalesced final store: out = logit - max - log(sum) ----
    for (int i = tid; i < BM * V; i += NTHR) {
        int r = i >> 7, v = i & (V - 1);
        out[(bBase + r) * V + v] = logit[r * LP + v] - rstat[r * 2] - rstat[r * 2 + 1];
    }
}

extern "C" void kernel_launch(void* const* d_in, const int* in_sizes, int n_in,
                              void* d_out, int out_size, void* d_ws, size_t ws_size,
                              hipStream_t stream) {
    const int*   seq   = (const int*)  d_in[0];
    const float* emb   = (const float*)d_in[1];
    const float* W_in  = (const float*)d_in[2];
    const float* b_in  = (const float*)d_in[3];
    const float* W_h   = (const float*)d_in[4];
    const float* b_h   = (const float*)d_in[5];
    const float* W_out = (const float*)d_in[6];
    const float* b_out = (const float*)d_in[7];
    float* out = (float*)d_out;

    const int nBlocks = Bsz / BM;   // 128
    rnn_char_model<<<nBlocks, NTHR, SMEM_BYTES, stream>>>(
        seq, emb, W_in, b_in, W_h, b_h, W_out, b_out, out);
}
